// Decoder_13606456394395
// MI455X (gfx1250) — matrix operand order repaired and checked by reading, not compiled, not run on hardware
//
#include <hip/hip_runtime.h>
#include <hip/hip_bf16.h>

typedef __attribute__((ext_vector_type(16))) __bf16 v16bf;
typedef __attribute__((ext_vector_type(8)))  float  v8f;

union Frag16 {
    v16bf v;
    uint4 q[2];
};

// ---------------------------------------------------------------------------
// NCHW f32 -> NHWC bf16
// ---------------------------------------------------------------------------
__global__ void nchw_to_nhwc_bf16(const float* __restrict__ src,
                                  __hip_bfloat16* __restrict__ dst,
                                  int B, int C, int H, int W) {
    int i = blockIdx.x * blockDim.x + threadIdx.x;
    int total = B * C * H * W;
    if (i >= total) return;
    int c = i % C;
    int r = i / C;
    int x = r % W; r /= W;
    int y = r % H;
    int b = r / H;
    dst[i] = __float2bfloat16(src[(((size_t)b * C + c) * H + y) * W + x]);
}

// ---------------------------------------------------------------------------
// Weight swizzle: w f32 [Co,Ci,3,3] (OIHW) ->
// bf16 [tap][ciChunk][coChunk][lane][16] laid out exactly as the 32x16
// 16-bit B-matrix WMMA fragment (lane L: column N = L%16; K-base = (L>=16)*8;
// element j within lane = K = kbase + j for j<8, 16 + kbase + (j-8) for j>=8).
// ---------------------------------------------------------------------------
__global__ void swizzle_weights(const float* __restrict__ w,
                                __hip_bfloat16* __restrict__ dst,
                                int Ci, int Co) {
    int i = blockIdx.x * blockDim.x + threadIdx.x;
    int total = 9 * Ci * Co;
    if (i >= total) return;
    int j    = i & 15;
    int r    = i >> 4;
    int lane = r & 31; r >>= 5;
    int nCo  = Co >> 4;
    int nCi  = Ci >> 5;
    int nc   = r % nCo; r /= nCo;
    int cc   = r % nCi;
    int tap  = r / nCi;
    int kbase = (lane >> 4) << 3;           // 0 or 8
    int ci = cc * 32 + kbase + j + ((j >> 3) << 3);
    int co = nc * 16 + (lane & 15);
    dst[i] = __float2bfloat16(w[((size_t)co * Ci + ci) * 9 + tap]);
}

// ---------------------------------------------------------------------------
// Style scales: s[b,co] = style[b,:] . fw[co,:] + fb[co]
// ---------------------------------------------------------------------------
__global__ void style_scales(const float* __restrict__ style,
                             const float* __restrict__ fw,
                             const float* __restrict__ fb,
                             float* __restrict__ s,
                             int B, int Co, int SD) {
    int i = blockIdx.x * blockDim.x + threadIdx.x;
    if (i >= B * Co) return;
    int b = i / Co, co = i % Co;
    float acc = fb[co];
    const float* sr = style + (size_t)b * SD;
    const float* wr = fw + (size_t)co * SD;
    for (int k = 0; k < SD; ++k) acc += sr[k] * wr[k];
    s[i] = acc;
}

// ---------------------------------------------------------------------------
// Main fused kernel: nearest-2x upsample + 3x3 conv (pad 1) + per-(b,co)
// scale + ReLU, as implicit GEMM on V_WMMA_F32_16X16X32_BF16.
// All shapes are compile-time template parameters (kills integer division
// and lets the K loop fully constant-fold).
//
//   in  : bf16 NHWC [B, Hin, Win, Ci]          (Ci multiple of 32)
//   wswz: bf16 [9][Ci/32][Co/16][32][16]       (pre-swizzled B fragments)
//   s   : f32  [B, Co]
//   out : bf16 NHWC [B, 2*Hin, 2*Win, Co]      (Co multiple of 64)
//
// One wave per block. Each wave computes 2 output rows x 16 columns x 64
// channels: 8 accumulators (2 M-tiles x 4 N-tiles), so each K-step issues
// 8 WMMAs against 4 A-loads + 8 B-loads (B fragments shared across rows).
// grid.x = B*(H/2)*(W/16), grid.y = Co/64.
// ---------------------------------------------------------------------------
template <int Hin, int Win, int Ci, int Co>
__global__ __launch_bounds__(32) void upconv_wmma(
    const __hip_bfloat16* __restrict__ in,
    const __hip_bfloat16* __restrict__ wswz,
    const float* __restrict__ s,
    __hip_bfloat16* __restrict__ out) {

    constexpr int H = Hin * 2, W = Win * 2;
    constexpr int nCi = Ci / 32;
    constexpr int nCo = Co / 16;
    constexpr int tilesPerRow = W / 16;
    constexpr int rowPairs = H / 2;

    const int lane = threadIdx.x;
    const int m  = lane & 15;           // A row within M-tile / D column
    const int hl = lane >> 4;           // half-lane select

    int t = blockIdx.x;
    const int b = t / (rowPairs * tilesPerRow);
    t -= b * (rowPairs * tilesPerRow);
    const int ry = t / tilesPerRow;
    const int xbase = (t - ry * tilesPerRow) * 16;
    const int y0 = ry * 2;
    const int x = xbase + m;
    const int ncBase = blockIdx.y * 4;

    v8f acc[2][4] = {};
    const int aOff = hl * 8;            // K offset of this lane's first b128

#pragma unroll
    for (int ky = 0; ky < 3; ++ky) {
#pragma unroll
        for (int kx = 0; kx < 3; ++kx) {
            const int xu = x + kx - 1;
            const bool xok = (xu >= 0) & (xu < W);
            const int ix = xu >> 1;     // nearest-2x: up[i] = in[i>>1]
            constexpr_helper: ;
            const int tap = ky * 3 + kx;
            const __hip_bfloat16* bbase = wswz + (size_t)tap * nCi * nCo * 512;

            const __hip_bfloat16* arow[2];
            bool ok[2];
#pragma unroll
            for (int mt = 0; mt < 2; ++mt) {
                const int yu = y0 + mt + ky - 1;
                ok[mt] = xok & (yu >= 0) & (yu < H);
                const int iy = yu >> 1;
                arow[mt] = in + ((size_t)((b * Hin + iy) * Win + ix)) * Ci + aOff;
            }

            for (int cc = 0; cc < nCi; ++cc) {
                // ---- B fragments (pre-swizzled, shared across row tiles) ----
                Frag16 bf_[4];
#pragma unroll
                for (int n = 0; n < 4; ++n) {
                    const uint4* bp = (const uint4*)(bbase +
                        (((size_t)cc * nCo + (ncBase + n)) * 32 + lane) * 16);
                    bf_[n].q[0] = bp[0];
                    bf_[n].q[1] = bp[1];
                }
#pragma unroll
                for (int mt = 0; mt < 2; ++mt) {
                    // ---- A fragment: two aligned b128 loads per lane ----
                    Frag16 af;
                    af.q[0] = make_uint4(0u, 0u, 0u, 0u);
                    af.q[1] = make_uint4(0u, 0u, 0u, 0u);
                    if (ok[mt]) {
                        const uint4* ap = (const uint4*)(arow[mt] + cc * 32);
                        af.q[0] = ap[0];   // K = aOff .. aOff+7
                        af.q[1] = ap[2];   // K = 16+aOff .. 16+aOff+7
                    }
#pragma unroll
                    for (int n = 0; n < 4; ++n) {
                        acc[mt][n] = __builtin_amdgcn_wmma_f32_16x16x32_bf16(
                            false, af.v, false, bf_[n].v,
                            (short)0, acc[mt][n], false, false);
                    }
                }
            }
        }
    }

    // Epilogue: D layout — lane holds column N = lane&15 (= out channel);
    // VGPR v holds row M = v + 8*hl (= x position). Scale + ReLU + bf16 store.
#pragma unroll
    for (int mt = 0; mt < 2; ++mt) {
        const int y = y0 + mt;
#pragma unroll
        for (int n = 0; n < 4; ++n) {
            const int co = (ncBase + n) * 16 + m;
            const float scv = s[b * Co + co];
#pragma unroll
            for (int v = 0; v < 8; ++v) {
                const int xm = xbase + v + hl * 8;
                float r = acc[mt][n][v] * scv;
                r = r > 0.0f ? r : 0.0f;
                out[((size_t)((b * H + y) * W + xm)) * Co + co] =
                    __float2bfloat16(r);
            }
        }
    }
}

// ---------------------------------------------------------------------------
// Final 3x3 conv 64 -> 3 with bias, bf16 NHWC in, f32 NCHW out.
// ---------------------------------------------------------------------------
__global__ void final_conv(const __hip_bfloat16* __restrict__ in,
                           const float* __restrict__ wf,
                           const float* __restrict__ bf,
                           float* __restrict__ out,
                           int B, int H, int W, int Ci) {
    int i = blockIdx.x * blockDim.x + threadIdx.x;
    if (i >= B * H * W) return;
    int x = i % W;
    int r = i / W;
    int y = r % H;
    int b = r / H;

    float a0 = bf[0], a1 = bf[1], a2 = bf[2];
    for (int ky = 0; ky < 3; ++ky) {
        int iy = y + ky - 1;
        if (iy < 0 || iy >= H) continue;
        for (int kx = 0; kx < 3; ++kx) {
            int ix = x + kx - 1;
            if (ix < 0 || ix >= W) continue;
            int tap = ky * 3 + kx;
            const __hip_bfloat16* p =
                in + ((size_t)((b * H + iy) * W + ix)) * Ci;
            for (int ci = 0; ci < Ci; ++ci) {
                float v = __bfloat162float(p[ci]);
                a0 += v * wf[((size_t)(0 * Ci + ci)) * 9 + tap];
                a1 += v * wf[((size_t)(1 * Ci + ci)) * 9 + tap];
                a2 += v * wf[((size_t)(2 * Ci + ci)) * 9 + tap];
            }
        }
    }
    size_t base = (size_t)b * 3 * H * W + (size_t)y * W + x;
    size_t hw = (size_t)H * W;
    out[base]          = a0;
    out[base + hw]     = a1;
    out[base + 2 * hw] = a2;
}

// ---------------------------------------------------------------------------
// Launch
// ---------------------------------------------------------------------------
static inline int cdiv(int a, int b) { return (a + b - 1) / b; }

extern "C" void kernel_launch(void* const* d_in, const int* in_sizes, int n_in,
                              void* d_out, int out_size, void* d_ws, size_t ws_size,
                              hipStream_t stream) {
    const float* x     = (const float*)d_in[0];   // [8,512,8,8]
    const float* style = (const float*)d_in[1];   // [8,512]
    const float* w[5]  = {(const float*)d_in[2],  (const float*)d_in[5],
                          (const float*)d_in[8],  (const float*)d_in[11],
                          (const float*)d_in[14]};
    const float* fw[5] = {(const float*)d_in[3],  (const float*)d_in[6],
                          (const float*)d_in[9],  (const float*)d_in[12],
                          (const float*)d_in[15]};
    const float* fb[5] = {(const float*)d_in[4],  (const float*)d_in[7],
                          (const float*)d_in[10], (const float*)d_in[13],
                          (const float*)d_in[16]};
    const float* wf  = (const float*)d_in[17];    // [3,64,3,3]
    const float* bfv = (const float*)d_in[18];    // [3]
    float* outp = (float*)d_out;                  // [8,3,256,256]

    const int B = 8, SD = 512;
    const int CiS[5] = {512, 256, 128, 64, 64};
    const int CoS[5] = {256, 128, 64, 64, 64};

    // ---- workspace carve-up ----
    char* p = (char*)d_ws;
    auto alloc = [&](size_t bytes) -> void* {
        void* r = (void*)p;
        p += (bytes + 255) & ~(size_t)255;
        return r;
    };
    __hip_bfloat16* xb   = (__hip_bfloat16*)alloc((size_t)B * 512 * 8 * 8 * 2);
    __hip_bfloat16* bufA = (__hip_bfloat16*)alloc((size_t)B * 256 * 256 * 64 * 2); // 64 MiB
    __hip_bfloat16* bufB = (__hip_bfloat16*)alloc((size_t)B * 128 * 128 * 64 * 2); // 16 MiB
    __hip_bfloat16* wz[5];
    float* sc[5];
    for (int i = 0; i < 5; ++i) {
        wz[i] = (__hip_bfloat16*)alloc((size_t)9 * CiS[i] * CoS[i] * 2);
        sc[i] = (float*)alloc((size_t)B * CoS[i] * 4);
    }

    // ---- prologue ----
    {
        int n = B * 512 * 8 * 8;
        nchw_to_nhwc_bf16<<<cdiv(n, 256), 256, 0, stream>>>(x, xb, B, 512, 8, 8);
    }
    for (int i = 0; i < 5; ++i) {
        int n = 9 * CiS[i] * CoS[i];
        swizzle_weights<<<cdiv(n, 256), 256, 0, stream>>>(w[i], wz[i], CiS[i], CoS[i]);
        int ns = B * CoS[i];
        style_scales<<<cdiv(ns, 128), 128, 0, stream>>>(style, fw[i], fb[i], sc[i],
                                                        B, CoS[i], SD);
    }

    // ---- 5 fused upsample+modconv+relu stages (ping-pong bf16 NHWC) ----
    // grid.x = B*(H/2)*(W/16), grid.y = Co/64
    upconv_wmma<8, 8, 512, 256>
        <<<dim3(B * 8 * 1, 4), 32, 0, stream>>>(xb,   wz[0], sc[0], bufA);
    upconv_wmma<16, 16, 256, 128>
        <<<dim3(B * 16 * 2, 2), 32, 0, stream>>>(bufA, wz[1], sc[1], bufB);
    upconv_wmma<32, 32, 128, 64>
        <<<dim3(B * 32 * 4, 1), 32, 0, stream>>>(bufB, wz[2], sc[2], bufA);
    upconv_wmma<64, 64, 64, 64>
        <<<dim3(B * 64 * 8, 1), 32, 0, stream>>>(bufA, wz[3], sc[3], bufB);
    upconv_wmma<128, 128, 64, 64>
        <<<dim3(B * 128 * 16, 1), 32, 0, stream>>>(bufB, wz[4], sc[4], bufA);

    // ---- final 64->3 conv, f32 NCHW output ----
    {
        int n = B * 256 * 256;
        final_conv<<<cdiv(n, 256), 256, 0, stream>>>(bufA, wf, bfv, outp,
                                                     B, 256, 256, 64);
    }
    (void)in_sizes; (void)n_in; (void)out_size; (void)ws_size;
}
